// SConvBlock_60730837565947
// MI455X (gfx1250) — matrix operand order repaired and checked by reading, not compiled, not run on hardware
//
#include <hip/hip_runtime.h>

#define BB   2
#define LL   2048
#define DD   1024
#define DFF  4096
#define ROWS (BB * LL)          // 4096
#define NE   (ROWS * DD)        // 4,194,304
#define NCH  16
#define CLEN (LL / NCH)         // 128

typedef __attribute__((ext_vector_type(16))) __bf16 v16bf;
typedef __attribute__((ext_vector_type(8)))  __bf16 v8bf;
typedef __attribute__((ext_vector_type(8)))  float  v8f;
typedef __attribute__((ext_vector_type(4)))  int    v4i;

#if defined(__HIP_DEVICE_COMPILE__) && \
    __has_builtin(__builtin_amdgcn_global_load_async_to_lds_b128) && \
    __has_builtin(__builtin_amdgcn_s_wait_asynccnt)
#define ASYNC_LDS 1
// builtin wants (int4 addrspace(1)*, int4 addrspace(3)*, imm offset, imm cpol)
#define AS1_V4I(p) ((__attribute__((address_space(1))) v4i*)(p))
#define AS3_V4I(p) ((__attribute__((address_space(3))) v4i*)(p))
#else
#define ASYNC_LDS 0
#endif

// ---------------------------------------------------------------- utilities

__device__ __forceinline__ float silu_f(float v) {
    return v / (1.0f + __expf(-v));
}

// two-value block reduction (sum, sumsq) for 256-thread blocks, wave32
__device__ __forceinline__ void block_reduce_2(float& s, float& s2, float* sh /*16 floats*/) {
    #pragma unroll
    for (int off = 16; off > 0; off >>= 1) {
        s  += __shfl_xor(s,  off, 32);
        s2 += __shfl_xor(s2, off, 32);
    }
    const int wid = threadIdx.x >> 5;
    __syncthreads();                 // protect sh against prior use
    if ((threadIdx.x & 31) == 0) { sh[wid] = s; sh[8 + wid] = s2; }
    __syncthreads();
    float a = 0.0f, b = 0.0f;
    #pragma unroll
    for (int i = 0; i < 8; ++i) { a += sh[i]; b += sh[8 + i]; }
    s = a; s2 = b;
}

// ---------------------------------------------------------------- f32 -> bf16

__global__ __launch_bounds__(256) void cvt_bf16_kernel(const float* __restrict__ src,
                                                       __bf16* __restrict__ dst, int n) {
    int i = blockIdx.x * 256 + threadIdx.x;
    if (i < n) dst[i] = (__bf16)src[i];
}

// ---------------------------------------------------------------- LayerNorm (1 block per row of D=1024)

__global__ __launch_bounds__(256) void ln_kernel(const float* __restrict__ in,
                                                 const float* __restrict__ w,
                                                 const float* __restrict__ bvec,
                                                 float* __restrict__ of,
                                                 __bf16* __restrict__ ob) {
    __shared__ float sh[16];
    const int row = blockIdx.x;
    const size_t base = (size_t)row * DD;
    float v[4];
    float s = 0.0f, s2 = 0.0f;
    #pragma unroll
    for (int i = 0; i < 4; ++i) {
        v[i] = in[base + threadIdx.x + i * 256];
        s += v[i]; s2 += v[i] * v[i];
    }
    block_reduce_2(s, s2, sh);
    const float mean = s * (1.0f / DD);
    const float var  = s2 * (1.0f / DD) - mean * mean;
    const float inv  = rsqrtf(var + 1e-5f);
    #pragma unroll
    for (int i = 0; i < 4; ++i) {
        const int d = threadIdx.x + i * 256;
        const float o = (v[i] - mean) * inv * w[d] + bvec[d];
        if (of) of[base + d] = o;
        if (ob) ob[base + d] = (__bf16)o;
    }
}

// ---------------------------------------------------------------- bf16 WMMA GEMM
// C[M,N] = epi( A[M,K] @ W[K,N] + bias[N] )   A,W bf16 row-major, f32 accumulate.
// Block: 256 threads = 8 waves, tile 128(M) x 128(N), K-step 32, double-buffered LDS.
// Wave layout: wm = wave>>1 (4 strips of 32 M-rows), wn = wave&1 (2 strips of 64 N-cols).
// Per wave per K-step: 2 A-fragments x 4 B-fragments = 8 WMMAs.

enum { EPI_NONE = 0, EPI_SILU = 1, EPI_SCALE = 2, EPI_SILU_BF16 = 3, EPI_RESID = 4 };

#define TM 128
#define TN 128
#define TK 32

template <int EPI>
__global__ __launch_bounds__(256) void gemm_bf16_kernel(const __bf16* __restrict__ A,
                                                        const __bf16* __restrict__ Wm,
                                                        const float* __restrict__ bias,
                                                        const float* __restrict__ resid,
                                                        float* __restrict__ Cf,
                                                        __bf16* __restrict__ Cb,
                                                        int Mdim, int Ndim, int Kdim) {
    __shared__ __bf16 As[2][TM][TK];   // 2 x 8 KB
    __shared__ __bf16 Bs[2][TN][TK];   // 2 x 8 KB, transposed W tile: Bs[n][k]
    const int n0   = blockIdx.x * TN;
    const int m0   = blockIdx.y * TM;
    const int t    = threadIdx.x;
    const int wave = t >> 5;
    const int lane = t & 31;
    const int lhalf = lane >> 4;    // 0 | 1
    const int l16   = lane & 15;
    const int wm    = wave >> 1;    // 0..3
    const int wn    = wave & 1;     // 0..1

    v8f acc[2][4];
    #pragma unroll
    for (int nt = 0; nt < 4; ++nt) {
        const float bv = bias[n0 + wn * 64 + nt * 16 + l16];
        #pragma unroll
        for (int i = 0; i < 8; ++i) { acc[0][nt][i] = bv; acc[1][nt][i] = bv; }
    }

    const int lr  = t >> 1;        // A-tile row this thread stages
    const int lhk = (t & 1) * 16;  // A-tile k-half
    const int wk  = t & 31;        // W-tile k this thread stages
    const int wng = t >> 5;        // W-tile n-group (16 cols)

    v8bf breg0, breg1;             // W staging registers (for LDS transpose)

    // stage A tile for K-offset k0 into buffer buf (async global->LDS when available)
    auto stageA = [&](int k0, int buf) {
        const __bf16* s = A + (size_t)(m0 + lr) * Kdim + k0 + lhk;
#if ASYNC_LDS
        __builtin_amdgcn_global_load_async_to_lds_b128(
            AS1_V4I(s),     AS3_V4I(&As[buf][lr][lhk]),     0, 0);
        __builtin_amdgcn_global_load_async_to_lds_b128(
            AS1_V4I(s + 8), AS3_V4I(&As[buf][lr][lhk + 8]), 0, 0);
#else
        *(v16bf*)&As[buf][lr][lhk] = *(const v16bf*)s;
#endif
    };
    auto loadB = [&](int k0) {
        const __bf16* s = Wm + (size_t)(k0 + wk) * Ndim + n0 + wng * 16;
        breg0 = *(const v8bf*)s;
        breg1 = *(const v8bf*)(s + 8);
    };
    auto storeB = [&](int buf) {
        #pragma unroll
        for (int j = 0; j < 8; ++j) {
            Bs[buf][wng * 16 + j][wk]     = breg0[j];
            Bs[buf][wng * 16 + 8 + j][wk] = breg1[j];
        }
    };

    // prologue: stage tile 0
    stageA(0, 0);
    loadB(0);
    storeB(0);
#if ASYNC_LDS
    __builtin_amdgcn_s_wait_asynccnt(0);
#endif
    __syncthreads();

    const int nk = Kdim / TK;
    for (int kt = 0; kt < nk; ++kt) {
        const int buf = kt & 1, nb = buf ^ 1;
        const bool more = (kt + 1 < nk);
        if (more) { stageA((kt + 1) * TK, nb); loadB((kt + 1) * TK); }

        // A fragments: lane<16 -> row=l16, k {0..15}; lane>=16 -> k {16..31}
        const int kb = lhalf * 16;
        v16bf afrag[2];
        #pragma unroll
        for (int am = 0; am < 2; ++am) {
            const int arow = wm * 32 + am * 16 + l16;
            const v8bf lo = *(const v8bf*)&As[buf][arow][kb];
            const v8bf hi = *(const v8bf*)&As[buf][arow][kb + 8];
            #pragma unroll
            for (int j = 0; j < 8; ++j) { afrag[am][j] = lo[j]; afrag[am][8 + j] = hi[j]; }
        }
        #pragma unroll
        for (int nt = 0; nt < 4; ++nt) {
            // B fragment: lane holds col, k = lhalf*16 .. +16 contiguous
            const v16bf bfrag = *(const v16bf*)&Bs[buf][wn * 64 + nt * 16 + l16][lhalf * 16];
            #pragma unroll
            for (int am = 0; am < 2; ++am) {
                acc[am][nt] = __builtin_amdgcn_wmma_f32_16x16x32_bf16(
                    false, afrag[am], false, bfrag, (short)0, acc[am][nt], false, false);
            }
        }
        if (more) storeB(nb);
#if ASYNC_LDS
        __builtin_amdgcn_s_wait_asynccnt(0);
#endif
        __syncthreads();
    }

    // epilogue: C element (m,n): lane = n%16 + 16*((m%16)>=8), vgpr = m%8
    #pragma unroll
    for (int am = 0; am < 2; ++am) {
        #pragma unroll
        for (int nt = 0; nt < 4; ++nt) {
            #pragma unroll
            for (int v = 0; v < 8; ++v) {
                const int m = m0 + wm * 32 + am * 16 + lhalf * 8 + v;
                const int n = n0 + wn * 64 + nt * 16 + l16;
                const size_t idx = (size_t)m * Ndim + n;
                float c = acc[am][nt][v];
                if (EPI == EPI_SILU || EPI == EPI_SILU_BF16) c = silu_f(c);
                if (EPI == EPI_SCALE)
                    c *= __expf(-6.90775527898f * ((float)n / (float)(Ndim - 1)));  // 0.001^(n/(N-1))
                if (EPI == EPI_RESID) c += resid[idx];
                if (EPI == EPI_SILU_BF16) Cb[idx] = (__bf16)c;
                else                      Cf[idx] = c;
            }
        }
    }
    (void)Mdim;
}

// ---------------------------------------------------------------- sconv scan
// p[l] = i*a[l] - 0.001 ; e^{p} = decay*(cos a + i sin a), decay = exp(-0.001)
// h[l] = e^{p[l]} h[l-1] + xn[l] ; g[l] = g[l-1] e^{p[l]} ; out = Re(h) + Re(g*lci)
// Chunked: phase1 stores per-chunk affine (P = prod e^p, H = local h from 0).

#define DECAY 0.999000499833375f

__global__ __launch_bounds__(256) void scan_phase1(const float* __restrict__ a,
                                                   const float* __restrict__ xn,
                                                   float* __restrict__ cs) {
    const int idx = blockIdx.x * 256 + threadIdx.x;     // (b, c, d)
    const int d = idx & (DD - 1);
    const int c = (idx / DD) & (NCH - 1);
    const int b = idx / (DD * NCH);
    float pr = 1.0f, pi = 0.0f, hr = 0.0f, hi = 0.0f;
    #pragma unroll 4
    for (int j = 0; j < CLEN; ++j) {
        const size_t off = ((size_t)b * LL + c * CLEN + j) * DD + d;
        const float av = a[off];
        float sn, csn; __sincosf(av, &sn, &csn);
        const float er = DECAY * csn, ei = DECAY * sn;
        const float x = xn[off];
        const float nhr = er * hr - ei * hi + x;
        const float nhi = er * hi + ei * hr;
        hr = nhr; hi = nhi;
        const float npr = er * pr - ei * pi;
        const float npi = er * pi + ei * pr;
        pr = npr; pi = npi;
    }
    float* p = cs + ((size_t)(b * DD + d) * NCH + c) * 4;
    p[0] = pr; p[1] = pi; p[2] = hr; p[3] = hi;
}

__global__ __launch_bounds__(256) void scan_phase2(const float* __restrict__ a,
                                                   const float* __restrict__ xn,
                                                   const float* __restrict__ cs,
                                                   const float* __restrict__ lci_re,
                                                   const float* __restrict__ lci_im,
                                                   float* __restrict__ hout) {
    const int idx = blockIdx.x * 256 + threadIdx.x;
    const int d = idx & (DD - 1);
    const int c = (idx / DD) & (NCH - 1);
    const int b = idx / (DD * NCH);
    // compose predecessor chunks: h = P*h + H ; g *= P
    float gr = 1.0f, gi = 0.0f, hr = 0.0f, hi = 0.0f;
    const float* base = cs + (size_t)(b * DD + d) * NCH * 4;
    for (int cc = 0; cc < c; ++cc) {
        const float Pr = base[cc * 4 + 0], Pi = base[cc * 4 + 1];
        const float Hr = base[cc * 4 + 2], Hi = base[cc * 4 + 3];
        const float nhr = Pr * hr - Pi * hi + Hr;
        const float nhi = Pr * hi + Pi * hr + Hi;
        hr = nhr; hi = nhi;
        const float ngr = Pr * gr - Pi * gi;
        const float ngi = Pr * gi + Pi * gr;
        gr = ngr; gi = ngi;
    }
    const float lr = lci_re[d], li = lci_im[d];
    #pragma unroll 4
    for (int j = 0; j < CLEN; ++j) {
        const size_t off = ((size_t)b * LL + c * CLEN + j) * DD + d;
        const float av = a[off];
        float sn, csn; __sincosf(av, &sn, &csn);
        const float er = DECAY * csn, ei = DECAY * sn;
        const float x = xn[off];
        const float nhr = er * hr - ei * hi + x;
        const float nhi = er * hi + ei * hr;
        hr = nhr; hi = nhi;
        const float ngr = er * gr - ei * gi;
        const float ngi = er * gi + ei * gr;
        gr = ngr; gi = ngi;
        hout[off] = hr + gr * lr - gi * li;
    }
}

// ---------------------------------------------------------------- mid fusion:
// s = LN(h)*w_in+b_in ; x2 = s*y + x ; xn2 = LN(x2)*w_ffn+b_ffn (bf16)

__global__ __launch_bounds__(256) void fuse_mid_kernel(const float* __restrict__ h,
                                                       const float* __restrict__ y,
                                                       const float* __restrict__ x,
                                                       const float* __restrict__ wi,
                                                       const float* __restrict__ bi,
                                                       const float* __restrict__ wf,
                                                       const float* __restrict__ bf,
                                                       float* __restrict__ x2,
                                                       __bf16* __restrict__ xn2) {
    __shared__ float sh[16];
    const size_t base = (size_t)blockIdx.x * DD;
    float hv[4];
    float s = 0.0f, s2 = 0.0f;
    #pragma unroll
    for (int i = 0; i < 4; ++i) {
        hv[i] = h[base + threadIdx.x + i * 256];
        s += hv[i]; s2 += hv[i] * hv[i];
    }
    block_reduce_2(s, s2, sh);
    float mean = s * (1.0f / DD);
    float inv  = rsqrtf(s2 * (1.0f / DD) - mean * mean + 1e-5f);
    float xv[4];
    float t1 = 0.0f, t2 = 0.0f;
    #pragma unroll
    for (int i = 0; i < 4; ++i) {
        const int d = threadIdx.x + i * 256;
        const float sv = (hv[i] - mean) * inv * wi[d] + bi[d];
        const float v  = sv * y[base + d] + x[base + d];
        xv[i] = v;
        x2[base + d] = v;
        t1 += v; t2 += v * v;
    }
    block_reduce_2(t1, t2, sh);
    mean = t1 * (1.0f / DD);
    inv  = rsqrtf(t2 * (1.0f / DD) - mean * mean + 1e-5f);
    #pragma unroll
    for (int i = 0; i < 4; ++i) {
        const int d = threadIdx.x + i * 256;
        xn2[base + d] = (__bf16)((xv[i] - mean) * inv * wf[d] + bf[d]);
    }
}

// ---------------------------------------------------------------- launch

extern "C" void kernel_launch(void* const* d_in, const int* in_sizes, int n_in,
                              void* d_out, int out_size, void* d_ws, size_t ws_size,
                              hipStream_t stream) {
    const float* x       = (const float*)d_in[0];
    const float* ln_sc_w = (const float*)d_in[1];
    const float* ln_sc_b = (const float*)d_in[2];
    const float* fc_w    = (const float*)d_in[3];
    const float* fc_b    = (const float*)d_in[4];
    const float* phz_w   = (const float*)d_in[5];
    const float* phz_b   = (const float*)d_in[6];
    const float* lci_re  = (const float*)d_in[7];
    const float* lci_im  = (const float*)d_in[8];
    const float* ln_in_w = (const float*)d_in[9];
    const float* ln_in_b = (const float*)d_in[10];
    const float* ln_ffn_w= (const float*)d_in[11];
    const float* ln_ffn_b= (const float*)d_in[12];
    const float* fc1_w   = (const float*)d_in[13];
    const float* fc1_b   = (const float*)d_in[14];
    const float* fc2_w   = (const float*)d_in[15];
    const float* fc2_b   = (const float*)d_in[16];
    float* out = (float*)d_out;

    // workspace carve
    char* wp = (char*)d_ws;
    auto carve = [&](size_t bytes) -> char* {
        char* p = wp; wp += (bytes + 255) & ~(size_t)255; return p;
    };
    float*  xn_f   = (float*)carve((size_t)NE * 4);
    float*  y_f    = (float*)carve((size_t)NE * 4);
    float*  a_f    = (float*)carve((size_t)NE * 4);
    float*  h_f    = (float*)carve((size_t)NE * 4);
    float*  x2_f   = (float*)carve((size_t)NE * 4);
    __bf16* xn_b   = (__bf16*)carve((size_t)NE * 2);
    __bf16* xn2_b  = (__bf16*)carve((size_t)NE * 2);
    __bf16* ff1_b  = (__bf16*)carve((size_t)ROWS * DFF * 2);
    __bf16* wfc_b  = (__bf16*)carve((size_t)DD * DD * 2);
    __bf16* wphz_b = (__bf16*)carve((size_t)DD * DD * 2);
    __bf16* wfc1_b = (__bf16*)carve((size_t)DD * DFF * 2);
    __bf16* wfc2_b = (__bf16*)carve((size_t)DFF * DD * 2);
    float*  cs     = (float*)carve((size_t)BB * DD * NCH * 4 * 4);

    // 1) weight conversion to bf16
    cvt_bf16_kernel<<<(DD * DD) / 256, 256, 0, stream>>>(fc_w,  wfc_b,  DD * DD);
    cvt_bf16_kernel<<<(DD * DD) / 256, 256, 0, stream>>>(phz_w, wphz_b, DD * DD);
    cvt_bf16_kernel<<<(DD * DFF) / 256, 256, 0, stream>>>(fc1_w, wfc1_b, DD * DFF);
    cvt_bf16_kernel<<<(DFF * DD) / 256, 256, 0, stream>>>(fc2_w, wfc2_b, DFF * DD);

    // 2) xn = LN(x)
    ln_kernel<<<ROWS, 256, 0, stream>>>(x, ln_sc_w, ln_sc_b, xn_f, xn_b);

    // 3) y = silu(xn @ fc_w + fc_b)
    {
        dim3 g(DD / TN, ROWS / TM);
        gemm_bf16_kernel<EPI_SILU><<<g, 256, 0, stream>>>(
            xn_b, wfc_b, fc_b, nullptr, y_f, nullptr, ROWS, DD, DD);
    }
    // 4) a = (xn @ phz_w + phz_b) * scale
    {
        dim3 g(DD / TN, ROWS / TM);
        gemm_bf16_kernel<EPI_SCALE><<<g, 256, 0, stream>>>(
            xn_b, wphz_b, phz_b, nullptr, a_f, nullptr, ROWS, DD, DD);
    }
    // 5) chunked complex scan -> h.real
    scan_phase1<<<(BB * NCH * DD) / 256, 256, 0, stream>>>(a_f, xn_f, cs);
    scan_phase2<<<(BB * NCH * DD) / 256, 256, 0, stream>>>(a_f, xn_f, cs, lci_re, lci_im, h_f);

    // 6) x2 = LN_in(h)*y + x ; xn2 = LN_ffn(x2)
    fuse_mid_kernel<<<ROWS, 256, 0, stream>>>(h_f, y_f, x, ln_in_w, ln_in_b,
                                              ln_ffn_w, ln_ffn_b, x2_f, xn2_b);

    // 7) ff1 = silu(xn2 @ fc1_w + fc1_b) (bf16)
    {
        dim3 g(DFF / TN, ROWS / TM);
        gemm_bf16_kernel<EPI_SILU_BF16><<<g, 256, 0, stream>>>(
            xn2_b, wfc1_b, fc1_b, nullptr, nullptr, ff1_b, ROWS, DFF, DD);
    }
    // 8) out = x2 + ff1 @ fc2_w + fc2_b
    {
        dim3 g(DD / TN, ROWS / TM);
        gemm_bf16_kernel<EPI_RESID><<<g, 256, 0, stream>>>(
            ff1_b, wfc2_b, fc2_b, x2_f, out, nullptr, ROWS, DD, DFF);
    }
    (void)in_sizes; (void)n_in; (void)out_size; (void)ws_size;
}
